// MinkEncoderDecoderModified_40699110096956
// MI455X (gfx1250) — matrix operand order repaired and checked by reading, not compiled, not run on hardware
//
#include <hip/hip_runtime.h>
#include <hip/hip_bf16.h>
#include <vector>

typedef __bf16 bf16;
typedef bf16 v16bf __attribute__((ext_vector_type(16)));
typedef bf16 v8bf  __attribute__((ext_vector_type(8)));
typedef float v8f  __attribute__((ext_vector_type(8)));
typedef unsigned int v8u __attribute__((ext_vector_type(8)));

#define RSQRT_1PEPS 0.9999950000374997f   // 1/sqrt(1+1e-5)

// ------------------------------------------------------------------
// Elementwise helper kernels
// ------------------------------------------------------------------

// NCDHW f32 -> NDHWC bf16
__global__ __launch_bounds__(256)
void k_cvt_nchw_ndhwc(const float* __restrict__ in, bf16* __restrict__ out, int C, long NV) {
  long i = (long)blockIdx.x * 256 + threadIdx.x;
  long tot = NV * (long)C;
  if (i >= tot) return;
  long v = i / C; int c = (int)(i % C);
  out[i] = (bf16)in[(long)c * NV + v];
}

// Pack f32 weights (Co,Ci,T taps contiguous) into the exact WMMA A-fragment order:
//   [mtile][kblock][lane][16 bf16]   (each lane's fragment is one contiguous 32B run;
//   a wave's A fragment for one k-block is a fully coalesced 1KB block)
// lane<16 holds row co=mt*16+lane, K {0..15} of the block; lane>=16 same row M, K {16..31}.
// flip=1 reverses all spatial taps (for the ks=2 transposed conv).
__global__ __launch_bounds__(256)
void k_pack_w(const float* __restrict__ w, bf16* __restrict__ o,
              int Co, int CoPad, int Ci, int T, int flip) {
  long i = (long)blockIdx.x * 256 + threadIdx.x;
  long tot = (long)CoPad * Ci * T;
  if (i >= tot) return;
  const int  e    = (int)(i & 15);
  const long t0   = i >> 4;
  const int  lane = (int)(t0 & 31);
  const long t1   = t0 >> 5;
  const int  KB   = (Ci * T) >> 5;          // 32-wide K blocks
  const int  kb   = (int)(t1 % KB);
  const int  mt   = (int)(t1 / KB);
  const int  co   = mt * 16 + (lane & 15);
  const int  k    = kb * 32 + ((lane >> 4) ? 16 : 0) + e;
  const int  t    = k / Ci, ci = k % Ci;
  const int  ts   = flip ? (T - 1 - t) : t;
  float v = (co < Co) ? w[((long)co * Ci + ci) * T + ts] : 0.f;
  o[i] = (bf16)v;
}

// copy skip tensor (NDHWC, Cs channels) into concat buffer channels [coff, coff+Cs)
__global__ __launch_bounds__(256)
void k_copy_cat(const bf16* __restrict__ in, bf16* __restrict__ out,
                int Cs, int Ctot, int coff, long NV) {
  long i = (long)blockIdx.x * 256 + threadIdx.x;
  long tot = NV * (long)Cs;
  if (i >= tot) return;
  long v = i / Cs; int c = (int)(i % Cs);
  out[v * Ctot + coff + c] = in[i];
}

// ------------------------------------------------------------------
// Implicit-GEMM conv on V_WMMA_F32_16X16X32_BF16
// ------------------------------------------------------------------

struct CArgs {
  const bf16* in; const bf16* wpk; bf16* out; float* outf;
  const float* gam; const float* bet; const bf16* resid;
  int Di, Hi, Wi, Ci, Do, Ho, Wo, Co, CoPad;
  int str, pad, act /*0 none,1 relu,2 leaky*/, ocs;
  int lWi, lHi, lWo, lHo;   // log2 of Wi,Hi,Wo,Ho (all dims are powers of two)
  long NVo;
};

// Each wave computes a 16(Co) x 64(voxel) tile: one A fragment feeds 4 WMMAs
// (4 independent accumulate chains hide XDL latency; address VALU co-executes).
// Out-of-range conv taps use clamped (valid) addresses + VALU AND-mask zeroing,
// keeping EXEC untouched in the hot loop (WMMA requires EXEC all-ones anyway).
// MODE 0 = forward conv (stride/pad); MODE 1 = ks2-stride2 transposed conv with
// one parity tap per blockIdx.y (each output voxel depends on exactly one input voxel).
template <int MODE, int KS>
__global__ __launch_bounds__(256)
void k_conv_wmma(CArgs a) {
  const bf16* __restrict__ inp = a.in;
  const bf16* __restrict__ wpk = a.wpk;

  const int lane = threadIdx.x & 31;
  const int wv   = threadIdx.x >> 5;
  const int hi   = lane >> 4;
  const int col  = lane & 15;
  const int MT   = a.CoPad >> 4;
  const int cols = MODE ? (a.Di * a.Hi * a.Wi) : (int)a.NVo;
  const int NG   = (cols + 63) >> 6;                 // 64-voxel groups
  const int tile = (int)blockIdx.x * 8 + wv;
  if (tile >= MT * NG) return;
  const int mt = tile % MT;
  const int ng = tile / MT;

  // decompose the 4 column voxels of this wave (pow2 dims -> shifts/masks)
  const int Wm = (MODE ? a.Wi : a.Wo) - 1;
  const int Hm = (MODE ? a.Hi : a.Ho) - 1;
  const int lW = MODE ? a.lWi : a.lWo;
  const int lH = MODE ? a.lHi : a.lHo;
  int ww[4], hh[4], dd[4], vC[4]; bool jok[4];
#pragma unroll
  for (int j = 0; j < 4; j++) {
    int v = ng * 64 + j * 16 + col;
    jok[j] = v < cols;                 // wave-uniform (cols is a multiple of 16)
    int vc = jok[j] ? v : cols - 1;
    vC[j] = vc;
    ww[j] = vc & Wm;
    int t1 = vc >> lW;
    hh[j] = t1 & Hm;
    dd[j] = t1 >> lH;
  }

  const int KC = a.Ci >> 5;
  constexpr int T = KS * KS * KS;
  const int KB = (MODE ? 8 * a.Ci : T * a.Ci) >> 5;
  // A fragments in pre-swizzled order: one 32B contiguous load per lane per k-block
  const bf16* __restrict__ abase = wpk + ((size_t)mt * KB * 32 + lane) * 16;
  const int bsel = hi ? 16 : 0;

  const v8f z = {0.f, 0.f, 0.f, 0.f, 0.f, 0.f, 0.f, 0.f};
  v8f acc[4] = {z, z, z, z};
  union BU { v16bf v; v8u u; };

  if (MODE) {
    const int tap = (int)blockIdx.y;
    const bf16* __restrict__ ap = abase + (size_t)(tap * KC) * 512;
    size_t ib[4];
#pragma unroll
    for (int j = 0; j < 4; j++)
      ib[j] = ((((size_t)dd[j] << a.lHi) + hh[j] << a.lWi) + ww[j]) * a.Ci + bsel;
    for (int c = 0; c < KC; c++) {
      v16bf av = *(const v16bf*)(ap + (size_t)c * 512);
#pragma unroll
      for (int j = 0; j < 4; j++) {
        v16bf bv = *(const v16bf*)(inp + ib[j] + (size_t)c * 32);
        acc[j] = __builtin_amdgcn_wmma_f32_16x16x32_bf16(false, av, false, bv, (short)0,
                                                         acc[j], false, false);
      }
    }
  } else {
#pragma unroll 1
    for (int t = 0; t < T; t++) {
      const int kd = t / (KS * KS), kh = (t / KS) % KS, kw = t % KS;
      size_t ib[4]; unsigned msk[4];
#pragma unroll
      for (int j = 0; j < 4; j++) {
        int id = dd[j] * a.str + kd - a.pad;
        int ih = hh[j] * a.str + kh - a.pad;
        int iw = ww[j] * a.str + kw - a.pad;
        const bool ok = (unsigned)id < (unsigned)a.Di && (unsigned)ih < (unsigned)a.Hi &&
                        (unsigned)iw < (unsigned)a.Wi;
        msk[j] = ok ? 0xFFFFFFFFu : 0u;
        if (!ok) { id = 0; ih = 0; iw = 0; }   // clamped, always-valid address
        ib[j] = ((((size_t)id << a.lHi) + ih << a.lWi) + iw) * a.Ci + bsel;
      }
      const bf16* __restrict__ ap = abase + (size_t)(t * KC) * 512;
      for (int c = 0; c < KC; c++) {
        v16bf av = *(const v16bf*)(ap + (size_t)c * 512);
#pragma unroll
        for (int j = 0; j < 4; j++) {
          BU b; b.v = *(const v16bf*)(inp + ib[j] + (size_t)c * 32);
          b.u = b.u & msk[j];                  // VALU zero-mask co-executes with WMMA
          acc[j] = __builtin_amdgcn_wmma_f32_16x16x32_bf16(false, av, false, b.v, (short)0,
                                                           acc[j], false, false);
        }
      }
    }
  }

  // Epilogue: BN -> residual add -> activation -> store (8 consecutive channels/lane)
  const int co0 = mt * 16 + hi * 8;
  float sc[8], bs[8];
  if (a.gam) {
#pragma unroll
    for (int r = 0; r < 8; r++) { sc[r] = a.gam[co0 + r] * RSQRT_1PEPS; bs[r] = a.bet[co0 + r]; }
  } else {
#pragma unroll
    for (int r = 0; r < 8; r++) { sc[r] = 1.f; bs[r] = 0.f; }
  }
#pragma unroll
  for (int j = 0; j < 4; j++) {
    if (!jok[j]) continue;
    float vals[8];
#pragma unroll
    for (int r = 0; r < 8; r++) vals[r] = acc[j][r] * sc[r] + bs[r];
    if (a.resid) {
      v8bf rv = *(const v8bf*)(a.resid + (size_t)vC[j] * a.Co + co0);
#pragma unroll
      for (int r = 0; r < 8; r++) vals[r] += (float)rv[r];
    }
    if (a.act == 1) {
#pragma unroll
      for (int r = 0; r < 8; r++) vals[r] = fmaxf(vals[r], 0.f);
    } else if (a.act == 2) {
#pragma unroll
      for (int r = 0; r < 8; r++) vals[r] = vals[r] > 0.f ? vals[r] : 0.01f * vals[r];
    }
    if (a.outf) {  // head: f32 NCDHW, mask padded rows
#pragma unroll
      for (int r = 0; r < 8; r++)
        if (co0 + r < a.Co) a.outf[(size_t)(co0 + r) * a.NVo + vC[j]] = vals[r];
    } else {
      size_t ob;
      if (MODE) {
        const int tap = (int)blockIdx.y;
        const int od = 2 * dd[j] + (tap >> 2), oh = 2 * hh[j] + ((tap >> 1) & 1),
                  ow = 2 * ww[j] + (tap & 1);
        ob = ((((size_t)od << a.lHo) + oh << a.lWo) + ow) * a.ocs + co0;
      } else {
        ob = (size_t)vC[j] * a.ocs + co0;
      }
      v8bf ov;
#pragma unroll
      for (int r = 0; r < 8; r++) ov[r] = (bf16)vals[r];
      *(v8bf*)(a.out + ob) = ov;
    }
  }
}

// ------------------------------------------------------------------
// Host side
// ------------------------------------------------------------------

struct BlockCfg { int bcin, bcout, r1in, r1out, r2; };
static const BlockCfg BLK[8] = {
  {32, 32, 32, 32, 32}, {32, 32, 32, 64, 64}, {64, 64, 64, 128, 128}, {128, 128, 128, 256, 256},
  {256, 256, 384, 256, 256}, {256, 128, 192, 128, 128}, {128, 96, 128, 96, 96}, {96, 96, 128, 96, 96}
};
static const int DVv[5] = {64, 32, 16, 8, 4}, HVv[5] = {64, 32, 16, 8, 4}, WVv[5] = {32, 16, 8, 4, 2};

static int ilog2i(int x) { int l = 0; while ((1 << l) < x) l++; return l; }

struct Seq { std::vector<int> slot; std::vector<long> sz; };
static void seqAdd(Seq& s, int slot, long size) { s.slot.push_back(slot); s.sz.push_back(size); }

static void seqBlock(Seq& s, int b, bool sorted) {
  const BlockCfg& c = BLK[b];
  const int base = b * 18;
  const long bw = (long)c.bcout * c.bcin * 8;
  const bool ds = (c.r1in != c.r1out);
  const long w1 = (long)c.r1out * c.r1in * 27, w2 = (long)c.r1out * c.r1out * 27;
  const long wd = (long)c.r1out * c.r1in, rw = (long)c.r2 * c.r2 * 27;
  if (sorted) {  // dict keys sorted (jax pytree)
    seqAdd(s, base + 2, c.bcout); seqAdd(s, base + 1, c.bcout); seqAdd(s, base + 0, bw);
    seqAdd(s, base + 5, c.r1out); seqAdd(s, base + 8, c.r1out); if (ds) seqAdd(s, base + 11, c.r1out);
    seqAdd(s, base + 4, c.r1out); seqAdd(s, base + 7, c.r1out); if (ds) seqAdd(s, base + 10, c.r1out);
    seqAdd(s, base + 3, w1); seqAdd(s, base + 6, w2); if (ds) seqAdd(s, base + 9, wd);
    seqAdd(s, base + 14, c.r2); seqAdd(s, base + 17, c.r2);
    seqAdd(s, base + 13, c.r2); seqAdd(s, base + 16, c.r2);
    seqAdd(s, base + 12, rw);   seqAdd(s, base + 15, rw);
  } else {       // insertion order
    seqAdd(s, base + 0, bw); seqAdd(s, base + 1, c.bcout); seqAdd(s, base + 2, c.bcout);
    seqAdd(s, base + 3, w1); seqAdd(s, base + 4, c.r1out); seqAdd(s, base + 5, c.r1out);
    seqAdd(s, base + 6, w2); seqAdd(s, base + 7, c.r1out); seqAdd(s, base + 8, c.r1out);
    if (ds) { seqAdd(s, base + 9, wd); seqAdd(s, base + 10, c.r1out); seqAdd(s, base + 11, c.r1out); }
    seqAdd(s, base + 12, rw); seqAdd(s, base + 13, c.r2); seqAdd(s, base + 14, c.r2);
    seqAdd(s, base + 15, rw); seqAdd(s, base + 16, c.r2); seqAdd(s, base + 17, c.r2);
  }
}

static Seq buildCand(int which) {
  Seq s;
  const long XS = 32L * 64 * 64 * 32;
  const long HW = 20L * 96;
  switch (which) {
    case 0:  // x first, params insertion order, head last
      seqAdd(s, 145, XS); for (int b = 0; b < 8; b++) seqBlock(s, b, false); seqAdd(s, 144, HW); break;
    case 1:  // fully sorted pytree: params (head first, sorted leaves) then x
      seqAdd(s, 144, HW); for (int b = 0; b < 8; b++) seqBlock(s, b, true); seqAdd(s, 145, XS); break;
    case 2:  // x first, params sorted
      seqAdd(s, 145, XS); seqAdd(s, 144, HW); for (int b = 0; b < 8; b++) seqBlock(s, b, true); break;
    default: // params insertion, head, x last
      for (int b = 0; b < 8; b++) seqBlock(s, b, false); seqAdd(s, 144, HW); seqAdd(s, 145, XS); break;
  }
  return s;
}

static void run_pack(hipStream_t st, const float* w, bf16* o, int Co, int CoPad, int Ci, int T, int flip) {
  long tot = (long)CoPad * Ci * T;
  k_pack_w<<<(unsigned)((tot + 255) / 256), 256, 0, st>>>(w, o, Co, CoPad, Ci, T, flip);
}

static void run_conv(hipStream_t st, const bf16* in, const bf16* wpk, bf16* out, float* outf,
                     const float* g, const float* be, const bf16* resid,
                     int lin, int Ci, int lout, int Co, int CoPad,
                     int ks, int strd, int pad, int mode, int act, int ocs) {
  CArgs a;
  a.in = in; a.wpk = wpk; a.out = out; a.outf = outf; a.gam = g; a.bet = be; a.resid = resid;
  a.Di = DVv[lin]; a.Hi = HVv[lin]; a.Wi = WVv[lin]; a.Ci = Ci;
  a.Do = DVv[lout]; a.Ho = HVv[lout]; a.Wo = WVv[lout]; a.Co = Co; a.CoPad = CoPad;
  a.str = strd; a.pad = pad; a.act = act; a.ocs = ocs;
  a.lWi = ilog2i(a.Wi); a.lHi = ilog2i(a.Hi); a.lWo = ilog2i(a.Wo); a.lHo = ilog2i(a.Ho);
  a.NVo = (long)a.Do * a.Ho * a.Wo;
  long colsN = mode ? (long)a.Di * a.Hi * a.Wi : a.NVo;
  long tiles = (long)(CoPad / 16) * ((colsN + 63) / 64);
  dim3 grid((unsigned)((tiles + 7) / 8), mode ? 8u : 1u, 1u);
  if (mode)          k_conv_wmma<1, 2><<<grid, 256, 0, st>>>(a);
  else if (ks == 1)  k_conv_wmma<0, 1><<<grid, 256, 0, st>>>(a);
  else if (ks == 2)  k_conv_wmma<0, 2><<<grid, 256, 0, st>>>(a);
  else               k_conv_wmma<0, 3><<<grid, 256, 0, st>>>(a);
}

extern "C" void kernel_launch(void* const* d_in, const int* in_sizes, int n_in,
                              void* d_out, int out_size, void* d_ws, size_t ws_size,
                              hipStream_t stream) {
  (void)out_size; (void)ws_size;

  // ---- bind parameters (detect pytree flattening order from sizes) ----
  int idxMap[146]; for (int i = 0; i < 146; i++) idxMap[i] = -1;
  bool matched = false;
  for (int ci = 0; ci < 4 && !matched; ci++) {
    Seq s = buildCand(ci);
    if ((int)s.sz.size() != n_in) continue;
    bool ok = true;
    for (int i = 0; i < n_in; i++) if (s.sz[i] != (long)in_sizes[i]) { ok = false; break; }
    if (ok) { for (int i = 0; i < n_in; i++) idxMap[s.slot[i]] = i; matched = true; }
  }
  if (!matched) {
    Seq s = buildCand(0);
    for (size_t i = 0; i < s.slot.size() && i < (size_t)n_in; i++) idxMap[s.slot[i]] = (int)i;
  }
  auto P = [&](int slot) -> const float* {
    int k = idxMap[slot]; if (k < 0 || k >= n_in) k = 0;
    return (const float*)d_in[k];
  };

  // ---- workspace (bf16 NDHWC activations) ----
  const long NV0 = 131072, NV1 = 16384, NV2 = 2048, NV3 = 256, NV4 = 32;
  size_t off = 0;
  auto walloc = [&](size_t nelem) -> bf16* {
    bf16* p = (bf16*)((char*)d_ws + off);
    off = (off + nelem * sizeof(bf16) + 255) & ~(size_t)255;
    return p;
  };
  bf16* x_c  = walloc(NV0 * 32);
  bf16* x1   = walloc(NV1 * 32);
  bf16* x2   = walloc(NV2 * 64);
  bf16* x3   = walloc(NV3 * 128);
  bf16* x4   = walloc(NV4 * 256);
  bf16* Pcat = walloc(NV0 * 128);   // concat buffer / encoder bcb output
  bf16* Ptmp = walloc(NV0 * 96);
  bf16* Pskp = walloc(NV0 * 96);
  bf16* Pr1  = walloc(NV0 * 96);
  bf16* Pout0 = walloc(NV0 * 96);
  bf16* Pout1 = walloc(NV0 * 96);
  bf16* Wpk  = walloc((size_t)256 * 10368);  // largest packed weight (up1 res1 w1)

  // ---- input layout conversion ----
  {
    long tot = NV0 * 32;
    k_cvt_nchw_ndhwc<<<(unsigned)((tot + 255) / 256), 256, 0, stream>>>(P(145), x_c, 32, NV0);
  }

  // ---- encoder ----
  const bf16* cur = x_c;
  bf16* xout[4] = {x1, x2, x3, x4};
  for (int b = 0; b < 4; b++) {
    const BlockCfg& c = BLK[b];
    const int base = b * 18, lin = b, lout = b + 1;
    // BCB: 2x2x2 stride-2 conv, BN + LeakyReLU
    run_pack(stream, P(base + 0), Wpk, c.bcout, c.bcout, c.bcin, 8, 0);
    run_conv(stream, cur, Wpk, Pcat, nullptr, P(base + 1), P(base + 2), nullptr,
             lin, c.bcin, lout, c.bcout, c.bcout, 2, 2, 0, 0, 2, c.bcout);
    // res1 conv1: 3^3, BN + ReLU
    run_pack(stream, P(base + 3), Wpk, c.r1out, c.r1out, c.r1in, 27, 0);
    run_conv(stream, Pcat, Wpk, Ptmp, nullptr, P(base + 4), P(base + 5), nullptr,
             lout, c.r1in, lout, c.r1out, c.r1out, 3, 1, 1, 0, 1, c.r1out);
    // skip path
    const bf16* sb = Pcat;
    if (c.r1in != c.r1out) {
      run_pack(stream, P(base + 9), Wpk, c.r1out, c.r1out, c.r1in, 1, 0);
      run_conv(stream, Pcat, Wpk, Pskp, nullptr, P(base + 10), P(base + 11), nullptr,
               lout, c.r1in, lout, c.r1out, c.r1out, 1, 1, 0, 0, 0, c.r1out);
      sb = Pskp;
    }
    // res1 conv2: BN + skip-add + ReLU
    run_pack(stream, P(base + 6), Wpk, c.r1out, c.r1out, c.r1out, 27, 0);
    run_conv(stream, Ptmp, Wpk, Pr1, nullptr, P(base + 7), P(base + 8), sb,
             lout, c.r1out, lout, c.r1out, c.r1out, 3, 1, 1, 0, 1, c.r1out);
    // res2
    run_pack(stream, P(base + 12), Wpk, c.r2, c.r2, c.r2, 27, 0);
    run_conv(stream, Pr1, Wpk, Ptmp, nullptr, P(base + 13), P(base + 14), nullptr,
             lout, c.r2, lout, c.r2, c.r2, 3, 1, 1, 0, 1, c.r2);
    run_pack(stream, P(base + 15), Wpk, c.r2, c.r2, c.r2, 27, 0);
    run_conv(stream, Ptmp, Wpk, xout[b], nullptr, P(base + 16), P(base + 17), Pr1,
             lout, c.r2, lout, c.r2, c.r2, 3, 1, 1, 0, 1, c.r2);
    cur = xout[b];
  }

  // ---- decoder ----
  const bf16* skipsrc[4] = {x3, x2, x1, x_c};
  const int skipC[4] = {128, 64, 32, 32};
  bf16* pouts[2] = {Pout0, Pout1};
  for (int j = 0; j < 4; j++) {
    const int b = 4 + j; const BlockCfg& c = BLK[b];
    const int base = b * 18, lin = 4 - j, lout = 3 - j;
    const int Cd = c.bcout, Cs = skipC[j], Ctot = Cd + Cs;  // == c.r1in
    // DCB: 2x transposed conv (8 parity taps), BN + LeakyReLU, write into concat channels [0,Cd)
    run_pack(stream, P(base + 0), Wpk, Cd, Cd, c.bcin, 8, 1);
    run_conv(stream, cur, Wpk, Pcat, nullptr, P(base + 1), P(base + 2), nullptr,
             lin, c.bcin, lout, Cd, Cd, 2, 2, 0, 1, 2, Ctot);
    // concat: copy skip into channels [Cd, Ctot)
    long NVo = (long)DVv[lout] * HVv[lout] * WVv[lout];
    long tot = NVo * Cs;
    k_copy_cat<<<(unsigned)((tot + 255) / 256), 256, 0, stream>>>(skipsrc[j], Pcat, Cs, Ctot, Cd, NVo);
    // res1
    run_pack(stream, P(base + 3), Wpk, c.r1out, c.r1out, Ctot, 27, 0);
    run_conv(stream, Pcat, Wpk, Ptmp, nullptr, P(base + 4), P(base + 5), nullptr,
             lout, Ctot, lout, c.r1out, c.r1out, 3, 1, 1, 0, 1, c.r1out);
    run_pack(stream, P(base + 9), Wpk, c.r1out, c.r1out, Ctot, 1, 0);
    run_conv(stream, Pcat, Wpk, Pskp, nullptr, P(base + 10), P(base + 11), nullptr,
             lout, Ctot, lout, c.r1out, c.r1out, 1, 1, 0, 0, 0, c.r1out);
    run_pack(stream, P(base + 6), Wpk, c.r1out, c.r1out, c.r1out, 27, 0);
    run_conv(stream, Ptmp, Wpk, Pr1, nullptr, P(base + 7), P(base + 8), Pskp,
             lout, c.r1out, lout, c.r1out, c.r1out, 3, 1, 1, 0, 1, c.r1out);
    // res2
    run_pack(stream, P(base + 12), Wpk, c.r2, c.r2, c.r2, 27, 0);
    run_conv(stream, Pr1, Wpk, Ptmp, nullptr, P(base + 13), P(base + 14), nullptr,
             lout, c.r2, lout, c.r2, c.r2, 3, 1, 1, 0, 1, c.r2);
    run_pack(stream, P(base + 15), Wpk, c.r2, c.r2, c.r2, 27, 0);
    run_conv(stream, Ptmp, Wpk, pouts[j & 1], nullptr, P(base + 16), P(base + 17), Pr1,
             lout, c.r2, lout, c.r2, c.r2, 3, 1, 1, 0, 1, c.r2);
    cur = pouts[j & 1];
  }

  // ---- head: 1x1x1, 96 -> 20, f32 NCDHW straight into d_out ----
  run_pack(stream, P(144), Wpk, 20, 32, 96, 1, 0);
  run_conv(stream, cur, Wpk, nullptr, (float*)d_out, nullptr, nullptr, nullptr,
           0, 96, 0, 20, 32, 1, 1, 0, 0, 0, 32);
}
